// MyConv2D_44822278701238
// MI455X (gfx1250) — compile-verified, matrix operand order checked by reading, not run on hardware
//
#include <hip/hip_runtime.h>

typedef __attribute__((ext_vector_type(16))) _Float16 v16h;
typedef __attribute__((ext_vector_type(8)))  _Float16 v8h;
typedef __attribute__((ext_vector_type(4)))  _Float16 v4h;
typedef __attribute__((ext_vector_type(8)))  float    v8f;

#define NIMG 8
#define HH   128
#define WW   128
#define CC   64
#define HO   126
#define WO   126
#define FF   128
#define KK   576      // 3*3*64
#define LSTR 72       // 64 + 8 pad (f16 elems) -> 144B row stride, conflict-free

// ---------------------------------------------------------------- f32 -> f16
__global__ void cvt_f32_f16(const float* __restrict__ s,
                            _Float16* __restrict__ d, int n) {
  int i = (blockIdx.x * blockDim.x + threadIdx.x) * 4;
  if (i + 4 <= n) {
    float4 v = *(const float4*)(s + i);
    v4h h;
    h.x = (_Float16)v.x; h.y = (_Float16)v.y;
    h.z = (_Float16)v.z; h.w = (_Float16)v.w;
    *(v4h*)(d + i) = h;
  }
}

// Build a v16h WMMA fragment from LDS: two 16B ds loads (K pairs 0..7 and 16..23
// relative to the lane's half-offset), per ISA 7.12.2 16-bit A/B layout.
__device__ __forceinline__ v16h load_frag(const _Float16* p) {
  v8h lo = *(const v8h*)(p);
  v8h hi = *(const v8h*)(p + 16);
  return __builtin_shufflevector(lo, hi, 0,1,2,3,4,5,6,7,8,9,10,11,12,13,14,15);
}

// CDNA5 async copy: global -> LDS, no VGPR staging, tracked by ASYNCcnt.
// lds_off = LDS byte offset (low 32 bits of a generic shared pointer).
__device__ __forceinline__ void async_copy_b128(unsigned lds_off,
                                                const _Float16* gptr) {
  asm volatile("global_load_async_to_lds_b128 %0, %1, off"
               :: "v"(lds_off), "v"(gptr) : "memory");
}
__device__ __forceinline__ void wait_async0() {
  asm volatile("s_wait_asynccnt 0x0" ::: "memory");
}

// ------------------------------------------------------- im2col-GEMM w/ WMMA
__global__ __launch_bounds__(256)
void conv3x3_wmma(const _Float16* __restrict__ xh,  // (N,H,W,C) f16
                  const _Float16* __restrict__ wh,  // (F,K) f16
                  const float*    __restrict__ bias,
                  float*          __restrict__ out) {
  __shared__ _Float16 As[2][128 * LSTR];  // 128 pixels x 64 chans (padded)
  __shared__ _Float16 Bs[2][128 * LSTR];  // 128 filters x 64 k    (padded)

  const int tid  = threadIdx.x;
  const int lane = tid & 31;
  const int lrow = lane & 15;
  const int half = (lane >> 4) << 3;      // 0 or 8: K half-offset per layout
  const int wave = tid >> 5;              // 8 waves
  const int wm   = (wave & 3) * 32;       // wave's M (wo) base within tile
  const int wn   = (wave >> 2) * 64;      // wave's N (filter) base

  const int n  = blockIdx.x / HO;
  const int ho = blockIdx.x % HO;

  v8f acc[2][4];
  #pragma unroll
  for (int mi = 0; mi < 2; ++mi)
    #pragma unroll
    for (int ni = 0; ni < 4; ++ni)
      acc[mi][ni] = v8f{};

  const int XMAX = NIMG * HH * WW * CC - 8;   // clamp for right/bottom overrun

  // Issue async global->LDS copies for tap stage s (s = i*3 + j) into buf.
  // A source is contiguous: pixels wo..wo+127 of row (ho+i) at column offset
  // j form one 16KB run. 256 threads x 4 x 16B each for A and for B.
  auto async_stage = [&](int s, int buf) {
    const int i = s / 3, j = s % 3;
    const int abase = ((n * HH + ho + i) * WW + j) * CC;
    #pragma unroll
    for (int q = 0; q < 4; ++q) {
      const int c = tid + 256 * q;
      const int row = c >> 3, col = (c & 7) * 8;
      int idx = abase + c * 8;
      idx = idx > XMAX ? XMAX : idx;          // harmless clamp for guard lanes
      async_copy_b128((unsigned)(uintptr_t)&As[buf][row * LSTR + col],
                      xh + idx);
      async_copy_b128((unsigned)(uintptr_t)&Bs[buf][row * LSTR + col],
                      wh + row * KK + s * CC + col);
    }
  };

  async_stage(0, 0);
  wait_async0();
  __syncthreads();

  for (int s = 0; s < 9; ++s) {
    const int buf = s & 1;
    if (s < 8) async_stage(s + 1, buf ^ 1);   // overlap copy with math

    #pragma unroll
    for (int ks = 0; ks < 2; ++ks) {          // two K=32 steps per tap
      const int k0 = ks * 32 + half;
      v16h af[2], bf[4];
      #pragma unroll
      for (int mi = 0; mi < 2; ++mi)
        af[mi] = load_frag(&As[buf][(wm + mi * 16 + lrow) * LSTR + k0]);
      #pragma unroll
      for (int ni = 0; ni < 4; ++ni)
        bf[ni] = load_frag(&Bs[buf][(wn + ni * 16 + lrow) * LSTR + k0]);
      #pragma unroll
      for (int mi = 0; mi < 2; ++mi)
        #pragma unroll
        for (int ni = 0; ni < 4; ++ni)
          acc[mi][ni] = __builtin_amdgcn_wmma_f32_16x16x32_f16(
              false, af[mi], false, bf[ni], (short)0, acc[mi][ni],
              false, false);
    }

    if (s < 8) wait_async0();                 // copies for s+1 have landed
    __syncthreads();
  }

  // Epilogue: bias + store. D layout: elem r of v8f -> M = r + half, N = lrow.
  const int obase = (n * HO + ho) * WO;
  #pragma unroll
  for (int ni = 0; ni < 4; ++ni) {
    const int f  = wn + ni * 16 + lrow;
    const float bv = bias[f];
    #pragma unroll
    for (int mi = 0; mi < 2; ++mi) {
      #pragma unroll
      for (int r = 0; r < 8; ++r) {
        const int wo = wm + mi * 16 + half + r;
        if (wo < WO)
          out[(obase + wo) * FF + f] = acc[mi][ni][r] + bv;
      }
    }
  }
}

// --------------------------------------------------------------------- launch
extern "C" void kernel_launch(void* const* d_in, const int* in_sizes, int n_in,
                              void* d_out, int out_size, void* d_ws, size_t ws_size,
                              hipStream_t stream) {
  (void)in_sizes; (void)n_in; (void)out_size; (void)ws_size;
  const float* x = (const float*)d_in[0];   // (8,128,128,64)
  const float* w = (const float*)d_in[1];   // (128,576)
  const float* b = (const float*)d_in[2];   // (128,)
  float* out = (float*)d_out;               // (8,126,126,128)

  _Float16* xh = (_Float16*)d_ws;                         // 16.78 MB
  _Float16* wh = xh + (size_t)NIMG * HH * WW * CC;        // +144 KB

  const int nx = NIMG * HH * WW * CC;   // 8,388,608
  const int nw = FF * KK;               // 73,728
  cvt_f32_f16<<<nx / 1024, 256, 0, stream>>>(x, xh, nx);
  cvt_f32_f16<<<(nw + 1023) / 1024, 256, 0, stream>>>(w, wh, nw);

  conv3x3_wmma<<<NIMG * HO, 256, 0, stream>>>(xh, wh, b, out);
}
